// DeepersetAggr_45423574122646
// MI455X (gfx1250) — matrix operand order repaired and checked by reading, not compiled, not run on hardware
//
#include <hip/hip_runtime.h>
#include <hip/hip_bf16.h>

typedef __attribute__((ext_vector_type(16))) _Float16 v16h;
typedef __attribute__((ext_vector_type(8)))  float    v8f;

#define N_PTS   500000
#define BATCHES 16384
#define E_DIM   128
#define H_DIM   256
#define Y_DIM   64
#define LN_EPSF 1e-5f
#define NW      2       // waves (row-tiles) per workgroup
#define LDH     264     // padded LDS stride for the 16xH f32 tile

// ---------------------------------------------------------------------------
// Fragment loaders (CDNA5 WMMA 16x16x32 f16 layouts, ISA 7.12.2, wave32)
// ---------------------------------------------------------------------------

// B fragment: pre-packed contiguous 32B per lane.
__device__ __forceinline__ v16h load_b_frag(const _Float16* __restrict__ packed,
                                            int kf, int nt, int NT, int lane) {
  return ((const v16h*)packed)[(kf * NT + nt) * 32 + lane];
}

// A fragment from a row-major f32 matrix; rowp already points at this lane's row.
// Lane L: M = L&15 (handled by rowp), K = kbase + (v<4?0:16) + (L>>4)*8 + (v&3)*2 + p
__device__ __forceinline__ v16h load_a_f32(const float* __restrict__ rowp,
                                           int kbase, int lane) {
  int hh = (lane >> 4) * 8;
  v16h r;
#pragma unroll
  for (int v = 0; v < 8; ++v) {
    int K0 = kbase + ((v < 4) ? 0 : 16) + hh + (v & 3) * 2;
    float2 f = *(const float2*)(rowp + K0);
    r[2 * v]     = (_Float16)f.x;
    r[2 * v + 1] = (_Float16)f.y;
  }
  return r;
}

// A fragment from a row-major f16 matrix (per-lane row pointer, e.g. gathered).
__device__ __forceinline__ v16h load_a_f16(const _Float16* __restrict__ rowp,
                                           int kbase, int lane) {
  int hh = (lane >> 4) * 8;
  v16h r;
#pragma unroll
  for (int v = 0; v < 8; ++v) {
    int K0 = kbase + ((v < 4) ? 0 : 16) + hh + (v & 3) * 2;
    r[2 * v]     = rowp[K0];
    r[2 * v + 1] = rowp[K0 + 1];
  }
  return r;
}

// ---------------------------------------------------------------------------
// Utility kernels
// ---------------------------------------------------------------------------

__global__ void zero_f32(float* __restrict__ p, int n) {
  int i = blockIdx.x * blockDim.x + threadIdx.x;
  if (i < n) p[i] = 0.f;
}

__global__ void cvt_f32_to_f16(const float* __restrict__ src,
                               _Float16* __restrict__ dst, int n) {
  int i = blockIdx.x * blockDim.x + threadIdx.x;
  if (i < n) dst[i] = (_Float16)src[i];
}

// Pack a [K,N] f32 weight matrix into WMMA B-fragment order (f16).
// idx = ((kf*NT + nt)*32 + lane)*16 + e ; e = v*2+p
__global__ void pack_b(const float* __restrict__ src, _Float16* __restrict__ dst,
                       int K, int N) {
  int idx = blockIdx.x * blockDim.x + threadIdx.x;
  int total = K * N;
  if (idx >= total) return;
  int e    = idx & 15;
  int lane = (idx >> 4) & 31;
  int fi   = idx >> 9;
  int NT   = N >> 4;
  int nt   = fi % NT;
  int kf   = fi / NT;
  int v = e >> 1, p = e & 1;
  int n = nt * 16 + (lane & 15);
  int k = kf * 32 + ((v < 4) ? 0 : 16) + ((lane >> 4) * 8) + (v & 3) * 2 + p;
  dst[idx] = (_Float16)src[(size_t)k * N + n];
}

// ---------------------------------------------------------------------------
// Fused local MLP + segment-sum (atomic) kernel.
// One wave handles 16 rows. KF1 = Din/32 (4 => x only; 8 => concat(x, xgh[batch])).
// ---------------------------------------------------------------------------
template <int KF1>
__launch_bounds__(NW * 32)
__global__ void local_mlp_seg(const float*    __restrict__ x,
                              const _Float16* __restrict__ xgh,
                              const int*      __restrict__ batch,
                              const _Float16* __restrict__ w1p,
                              const float*    __restrict__ b1,
                              const float*    __restrict__ g,
                              const float*    __restrict__ bt,
                              const _Float16* __restrict__ w2p,
                              const float*    __restrict__ b2,
                              float*          __restrict__ sout) {
  __shared__ float t1[NW][16][LDH];
  __shared__ float ps[NW][32], pq[NW][32];
  __shared__ float mean_s[NW][16], rstd_s[NW][16];

  const int lane    = threadIdx.x & 31;
  const int w       = threadIdx.x >> 5;
  const int rowbase = (blockIdx.x * NW + w) * 16;
  const int m       = lane & 15;

  // A fragments for GEMM1
  v16h af[KF1];
  const float* xrow = x + (size_t)(rowbase + m) * E_DIM;
#pragma unroll
  for (int k = 0; k < 4; ++k) af[k] = load_a_f32(xrow, 32 * k, lane);
  if (KF1 == 8) {
    const _Float16* grow = xgh + (size_t)batch[rowbase + m] * E_DIM;
#pragma unroll
    for (int k = 0; k < 4; ++k) af[4 + k] = load_a_f16(grow, 32 * k, lane);
  }

  // GEMM1: t1 = x_ @ w1 + b1  (16 x H)
  for (int nt = 0; nt < H_DIM / 16; ++nt) {
    v8f acc = {};
#pragma unroll
    for (int k = 0; k < KF1; ++k) {
      v16h bf = load_b_frag(w1p, k, nt, H_DIM / 16, lane);
      acc = __builtin_amdgcn_wmma_f32_16x16x32_f16(false, af[k], false, bf,
                                                   (short)0, acc, false, false);
    }
    int n = nt * 16 + m;
    float bb = b1[n];
#pragma unroll
    for (int r = 0; r < 8; ++r)
      t1[w][r + ((lane >> 4) * 8)][n] = acc[r] + bb;
  }
  __syncthreads();

  // LayerNorm statistics: 2 lanes per row, each over H/2 columns
  {
    int hr = lane >> 4;
    float sum = 0.f, sq = 0.f;
    for (int c = 0; c < H_DIM / 2; ++c) {
      float v = t1[w][m][hr * (H_DIM / 2) + c];
      sum += v; sq += v * v;
    }
    ps[w][lane] = sum; pq[w][lane] = sq;
  }
  __syncthreads();
  if (lane < 16) {
    float sum  = ps[w][lane] + ps[w][lane + 16];
    float sq   = pq[w][lane] + pq[w][lane + 16];
    float mean = sum * (1.f / H_DIM);
    float var  = sq * (1.f / H_DIM) - mean * mean;
    mean_s[w][lane] = mean;
    rstd_s[w][lane] = rsqrtf(var + LN_EPSF);
  }
  __syncthreads();

  // A2 fragments: relu(LN(t1)*g + bt) as f16
  v16h a2[8];
  {
    float mn = mean_s[w][m], rs = rstd_s[w][m];
    int hh = (lane >> 4) * 8;
#pragma unroll
    for (int k = 0; k < 8; ++k) {
#pragma unroll
      for (int v = 0; v < 8; ++v) {
        int K0 = 32 * k + ((v < 4) ? 0 : 16) + hh + (v & 3) * 2;
#pragma unroll
        for (int p = 0; p < 2; ++p) {
          int Kc = K0 + p;
          float t = (t1[w][m][Kc] - mn) * rs * g[Kc] + bt[Kc];
          t = t > 0.f ? t : 0.f;
          a2[k][2 * v + p] = (_Float16)t;
        }
      }
    }
  }

  // GEMM2 + segment-sum atomics: s[batch[row]] += a2 @ w2 + b2
  for (int nt = 0; nt < E_DIM / 16; ++nt) {
    v8f acc = {};
#pragma unroll
    for (int k = 0; k < 8; ++k) {
      v16h bf = load_b_frag(w2p, k, nt, E_DIM / 16, lane);
      acc = __builtin_amdgcn_wmma_f32_16x16x32_f16(false, a2[k], false, bf,
                                                   (short)0, acc, false, false);
    }
    int n = nt * 16 + m;
    float bb = b2[n];
#pragma unroll
    for (int r = 0; r < 8; ++r) {
      int mm = r + ((lane >> 4) * 8);
      int bi = batch[rowbase + mm];
      atomicAdd(sout + (size_t)bi * E_DIM + n, acc[r] + bb);
    }
  }
}

// ---------------------------------------------------------------------------
// Fused global MLP over pooled rows; optionally adds y@cw + cb (layer 0).
// Writes f32 output and an f16 copy (for the next layer's gather).
// ---------------------------------------------------------------------------
__launch_bounds__(NW * 32)
__global__ void global_mlp(const float*    __restrict__ sin,
                           const _Float16* __restrict__ w1p,
                           const float*    __restrict__ b1,
                           const float*    __restrict__ g,
                           const float*    __restrict__ bt,
                           const _Float16* __restrict__ w2p,
                           const float*    __restrict__ b2,
                           const _Float16* __restrict__ yh,
                           const _Float16* __restrict__ cwp,
                           const float*    __restrict__ cb,
                           float*          __restrict__ outf,
                           _Float16*       __restrict__ outh,
                           int addy) {
  __shared__ float t1[NW][16][LDH];
  __shared__ float ps[NW][32], pq[NW][32];
  __shared__ float mean_s[NW][16], rstd_s[NW][16];

  const int lane    = threadIdx.x & 31;
  const int w       = threadIdx.x >> 5;
  const int rowbase = (blockIdx.x * NW + w) * 16;
  const int m       = lane & 15;

  v16h af[4];
  const float* srow = sin + (size_t)(rowbase + m) * E_DIM;
#pragma unroll
  for (int k = 0; k < 4; ++k) af[k] = load_a_f32(srow, 32 * k, lane);

  for (int nt = 0; nt < H_DIM / 16; ++nt) {
    v8f acc = {};
#pragma unroll
    for (int k = 0; k < 4; ++k) {
      v16h bf = load_b_frag(w1p, k, nt, H_DIM / 16, lane);
      acc = __builtin_amdgcn_wmma_f32_16x16x32_f16(false, af[k], false, bf,
                                                   (short)0, acc, false, false);
    }
    int n = nt * 16 + m;
    float bb = b1[n];
#pragma unroll
    for (int r = 0; r < 8; ++r)
      t1[w][r + ((lane >> 4) * 8)][n] = acc[r] + bb;
  }
  __syncthreads();

  {
    int hr = lane >> 4;
    float sum = 0.f, sq = 0.f;
    for (int c = 0; c < H_DIM / 2; ++c) {
      float v = t1[w][m][hr * (H_DIM / 2) + c];
      sum += v; sq += v * v;
    }
    ps[w][lane] = sum; pq[w][lane] = sq;
  }
  __syncthreads();
  if (lane < 16) {
    float sum  = ps[w][lane] + ps[w][lane + 16];
    float sq   = pq[w][lane] + pq[w][lane + 16];
    float mean = sum * (1.f / H_DIM);
    float var  = sq * (1.f / H_DIM) - mean * mean;
    mean_s[w][lane] = mean;
    rstd_s[w][lane] = rsqrtf(var + LN_EPSF);
  }
  __syncthreads();

  v16h a2[8];
  {
    float mn = mean_s[w][m], rs = rstd_s[w][m];
    int hh = (lane >> 4) * 8;
#pragma unroll
    for (int k = 0; k < 8; ++k) {
#pragma unroll
      for (int v = 0; v < 8; ++v) {
        int K0 = 32 * k + ((v < 4) ? 0 : 16) + hh + (v & 3) * 2;
#pragma unroll
        for (int p = 0; p < 2; ++p) {
          int Kc = K0 + p;
          float t = (t1[w][m][Kc] - mn) * rs * g[Kc] + bt[Kc];
          t = t > 0.f ? t : 0.f;
          a2[k][2 * v + p] = (_Float16)t;
        }
      }
    }
  }

  const _Float16* yrow = yh + (size_t)(rowbase + m) * Y_DIM;
  for (int nt = 0; nt < E_DIM / 16; ++nt) {
    v8f acc = {};
#pragma unroll
    for (int k = 0; k < 8; ++k) {
      v16h bf = load_b_frag(w2p, k, nt, E_DIM / 16, lane);
      acc = __builtin_amdgcn_wmma_f32_16x16x32_f16(false, a2[k], false, bf,
                                                   (short)0, acc, false, false);
    }
    float extra_b = 0.f;
    int n = nt * 16 + m;
    if (addy) {
#pragma unroll
      for (int k = 0; k < 2; ++k) {
        v16h afy = load_a_f16(yrow, 32 * k, lane);
        v16h bf  = load_b_frag(cwp, k, nt, E_DIM / 16, lane);
        acc = __builtin_amdgcn_wmma_f32_16x16x32_f16(false, afy, false, bf,
                                                     (short)0, acc, false, false);
      }
      extra_b = cb[n];
    }
    float bb = b2[n] + extra_b;
#pragma unroll
    for (int r = 0; r < 8; ++r) {
      int mm = r + ((lane >> 4) * 8);
      size_t o = (size_t)(rowbase + mm) * E_DIM + n;
      float val = acc[r] + bb;
      outf[o] = val;
      outh[o] = (_Float16)val;
    }
  }
}

// ---------------------------------------------------------------------------
// Host orchestration
// ---------------------------------------------------------------------------
extern "C" void kernel_launch(void* const* d_in, const int* in_sizes, int n_in,
                              void* d_out, int out_size, void* d_ws, size_t ws_size,
                              hipStream_t stream) {
  const float* x     = (const float*)d_in[0];
  const float* y     = (const float*)d_in[1];
  const int*   batch = (const int*)d_in[2];

  const float* l0_lw1 = (const float*)d_in[3];
  const float* l0_lb1 = (const float*)d_in[4];
  const float* l0_lg  = (const float*)d_in[5];
  const float* l0_lbt = (const float*)d_in[6];
  const float* l0_lw2 = (const float*)d_in[7];
  const float* l0_lb2 = (const float*)d_in[8];
  const float* l0_gw1 = (const float*)d_in[9];
  const float* l0_gb1 = (const float*)d_in[10];
  const float* l0_gg  = (const float*)d_in[11];
  const float* l0_gbt = (const float*)d_in[12];
  const float* l0_gw2 = (const float*)d_in[13];
  const float* l0_gb2 = (const float*)d_in[14];
  const float* lr_lw1 = (const float*)d_in[15];
  const float* lr_lb1 = (const float*)d_in[16];
  const float* lr_lg  = (const float*)d_in[17];
  const float* lr_lbt = (const float*)d_in[18];
  const float* lr_lw2 = (const float*)d_in[19];
  const float* lr_lb2 = (const float*)d_in[20];
  const float* lr_gw1 = (const float*)d_in[21];
  const float* lr_gb1 = (const float*)d_in[22];
  const float* lr_gg  = (const float*)d_in[23];
  const float* lr_gbt = (const float*)d_in[24];
  const float* lr_gw2 = (const float*)d_in[25];
  const float* lr_gb2 = (const float*)d_in[26];
  const float* cw     = (const float*)d_in[27];
  const float* cb     = (const float*)d_in[28];

  // ---- workspace layout ----
  char* ws = (char*)d_ws;
  size_t off = 0;
  auto alloc = [&](size_t bytes) {
    char* p = ws + off;
    off += (bytes + 255) & ~(size_t)255;
    return p;
  };
  float*    s_buf = (float*)alloc((size_t)BATCHES * E_DIM * 4);
  float*    xg    = (float*)alloc((size_t)BATCHES * E_DIM * 4);
  _Float16* xgh   = (_Float16*)alloc((size_t)BATCHES * E_DIM * 2);
  _Float16* yh    = (_Float16*)alloc((size_t)BATCHES * Y_DIM * 2);
  auto allocH = [&](size_t n) { return (_Float16*)alloc(n * 2); };
  _Float16* p_l0_lw1 = allocH(128 * 256);
  _Float16* p_l0_lw2 = allocH(256 * 128);
  _Float16* p_l0_gw1 = allocH(128 * 256);
  _Float16* p_l0_gw2 = allocH(256 * 128);
  _Float16* p_lr_lw1[2] = {allocH(256 * 256), allocH(256 * 256)};
  _Float16* p_lr_lw2[2] = {allocH(256 * 128), allocH(256 * 128)};
  _Float16* p_lr_gw1[2] = {allocH(128 * 256), allocH(128 * 256)};
  _Float16* p_lr_gw2[2] = {allocH(256 * 128), allocH(256 * 128)};
  _Float16* p_cw = allocH(64 * 128);
  (void)ws_size; (void)in_sizes; (void)n_in; (void)out_size;

  auto pack = [&](const float* src, _Float16* dst, int K, int N) {
    int total = K * N;
    pack_b<<<(total + 255) / 256, 256, 0, stream>>>(src, dst, K, N);
  };
  pack(l0_lw1, p_l0_lw1, 128, 256);
  pack(l0_lw2, p_l0_lw2, 256, 128);
  pack(l0_gw1, p_l0_gw1, 128, 256);
  pack(l0_gw2, p_l0_gw2, 256, 128);
  for (int i = 0; i < 2; ++i) {
    pack(lr_lw1 + (size_t)i * 256 * 256, p_lr_lw1[i], 256, 256);
    pack(lr_lw2 + (size_t)i * 256 * 128, p_lr_lw2[i], 256, 128);
    pack(lr_gw1 + (size_t)i * 128 * 256, p_lr_gw1[i], 128, 256);
    pack(lr_gw2 + (size_t)i * 256 * 128, p_lr_gw2[i], 256, 128);
  }
  pack(cw, p_cw, 64, 128);
  {
    int n = BATCHES * Y_DIM;
    cvt_f32_to_f16<<<(n + 255) / 256, 256, 0, stream>>>(y, yh, n);
  }

  const int SE       = BATCHES * E_DIM;
  const int zgrid    = (SE + 255) / 256;
  const int lgrid    = N_PTS / 16 / NW;     // 15625
  const int ggrid    = BATCHES / 16 / NW;   // 512
  const int threads  = NW * 32;

  // ---- layer 0 ----
  zero_f32<<<zgrid, 256, 0, stream>>>(s_buf, SE);
  local_mlp_seg<4><<<lgrid, threads, 0, stream>>>(
      x, (const _Float16*)nullptr, batch,
      p_l0_lw1, l0_lb1, l0_lg, l0_lbt, p_l0_lw2, l0_lb2, s_buf);
  global_mlp<<<ggrid, threads, 0, stream>>>(
      s_buf, p_l0_gw1, l0_gb1, l0_gg, l0_gbt, p_l0_gw2, l0_gb2,
      yh, p_cw, cb, xg, xgh, 1);

  // ---- layers 1..2 ----
  for (int i = 0; i < 2; ++i) {
    zero_f32<<<zgrid, 256, 0, stream>>>(s_buf, SE);
    local_mlp_seg<8><<<lgrid, threads, 0, stream>>>(
        x, xgh, batch,
        p_lr_lw1[i], lr_lb1 + i * H_DIM, lr_lg + i * H_DIM, lr_lbt + i * H_DIM,
        p_lr_lw2[i], lr_lb2 + i * E_DIM, s_buf);
    float* outf = (i == 1) ? (float*)d_out : xg;
    global_mlp<<<ggrid, threads, 0, stream>>>(
        s_buf, p_lr_gw1[i], lr_gb1 + i * H_DIM, lr_gg + i * H_DIM,
        lr_gbt + i * H_DIM, p_lr_gw2[i], lr_gb2 + i * E_DIM,
        yh, p_cw, cb, outf, xgh, 0);
  }
}
